// TokenDifferentialOperator_14637248545390
// MI455X (gfx1250) — compile-verified, matrix operand order checked by reading, not run on hardware
//
#include <hip/hip_runtime.h>

typedef __attribute__((ext_vector_type(16))) _Float16 v16h;
typedef __attribute__((ext_vector_type(8)))  _Float16 v8h;
typedef __attribute__((ext_vector_type(4)))  _Float16 v4h;
typedef __attribute__((ext_vector_type(8)))  float    v8f;

#define BH_     64
#define S_      4096
#define HD_     64
#define NF_     9
#define NSEG_   8
#define SEGROWS (S_ / NSEG_)   // 512
#define CH1_    256            // rows per chunk, kernel 1
#define ROWS2_  256            // rows per block, kernel 2

#define KSTRIDE 68             // floats per padded 64-wide f32 row (aligned 16B, bank-spread)
#define TSTR1   264            // halves per padded 256-wide f16 row (kernel1, transposed tiles)
#define ASTR2   72             // halves per padded 64-wide f16 row  (kernel2 A, row-major)
#define BSTR2   72             // halves per padded 64-wide f16 row  (kernel2 B = KV^T)

// LDS byte budgets (all 16B-aligned)
#define LDS_W    (128 * 16 * 4)                 // 8192
#define LDS_KF   (CH1_ * KSTRIDE * 4)           // 69632
#define LDS_T1   (64 * TSTR1 * 2)               // 33792
#define LDS1_SZ  (LDS_W + 2 * LDS_KF + 2 * LDS_T1)         // 215040
#define LDS_A2   (ROWS2_ * ASTR2 * 2)           // 36864
#define LDS_B2   (64 * BSTR2 * 2)               // 9216
#define LDS2_SZ  (LDS_W + 2 * LDS_KF + LDS_A2 + LDS_B2)    // 193536

// ---- WMMA fragment loader: matches CDNA5 16-bit A/B VGPR layout ----
// lanes 0-15 : row = r0+lane,     K = k0+{0..7} (lo b128) and k0+{16..23} (hi b128)
// lanes 16-31: row = r0+lane-16,  K = k0+{8..15}            and k0+{24..31}
__device__ __forceinline__ v16h load_frag(const _Float16* base, int stride,
                                          int row0, int k0, int lane) {
  const int row = row0 + (lane & 15);
  const int kh  = k0 + (lane >> 4) * 8;
  const _Float16* p = base + row * stride + kh;
  v8h lo = *(const v8h*)p;
  v8h hi = *(const v8h*)(p + 16);
  v16h r;
#pragma unroll
  for (int i = 0; i < 8; ++i) { r[i] = lo[i]; r[i + 8] = hi[i]; }
  return r;
}

__device__ __forceinline__ v8f wmma_f16(v16h a, v16h b, v8f c) {
  return __builtin_amdgcn_wmma_f32_16x16x32_f16(false, a, false, b,
                                                (short)0, c, false, false);
}

// ---- router: 9 scores over 128-dim [row|row'], STE-forward argmax -> lambda ----
// Wl is padded [128][16] in LDS so the unrolled n-loop vectorizes into ds_load_b128.
__device__ __forceinline__ float row_lambda(const float* krow, const float* prow,
                                            const float* Wl, const float* lamg) {
  float acc[NF_];
#pragma unroll
  for (int n = 0; n < NF_; ++n) acc[n] = 0.f;
  for (int d4 = 0; d4 < 16; ++d4) {
    float4 a = *(const float4*)(krow + d4 * 4);
    float4 b = *(const float4*)(prow + d4 * 4);
    float av[4] = {a.x, a.y, a.z, a.w};
    float bv[4] = {b.x, b.y, b.z, b.w};
#pragma unroll
    for (int j = 0; j < 4; ++j) {
      const int d = d4 * 4 + j;
      const float* w0 = Wl + d * 16;
      const float* w1 = Wl + (64 + d) * 16;
#pragma unroll
      for (int n = 0; n < NF_; ++n)
        acc[n] = __builtin_fmaf(av[j], w0[n], __builtin_fmaf(bv[j], w1[n], acc[n]));
    }
  }
  int best = 0; float bmax = acc[0];
#pragma unroll
  for (int n = 1; n < NF_; ++n)           // strict '>' keeps first max (jnp.argmax)
    if (acc[n] > bmax) { bmax = acc[n]; best = n; }
  return lamg[best];
}

extern "C" __global__ void td_zero_kernel(float* __restrict__ p, int n) {
  int i = blockIdx.x * blockDim.x + threadIdx.x;
  if (i < n) p[i] = 0.f;
}

// ================= Kernel 1: KV[b,h] = (K_t - lam_k K'_t)^T * V =================
extern "C" __global__ __launch_bounds__(256)
void td_kv_kernel(const float* __restrict__ Kt, const float* __restrict__ Kp,
                  const float* __restrict__ V,  const float* __restrict__ lamg,
                  const float* __restrict__ Wk, float* __restrict__ KV) {
  extern __shared__ char smem[];
  float*    Wl  = (float*)smem;
  float*    Ktf = (float*)(smem + LDS_W);
  float*    Kpf = (float*)(smem + LDS_W + LDS_KF);
  _Float16* At  = (_Float16*)(smem + LDS_W + 2 * LDS_KF);            // K_diff^T [d][s]
  _Float16* Bt  = (_Float16*)(smem + LDS_W + 2 * LDS_KF + LDS_T1);   // V^T      [e][s]

  const int tid  = threadIdx.x;
  const int lane = tid & 31;
  const int w    = tid >> 5;
  const int bh   = blockIdx.x / NSEG_;
  const int seg  = blockIdx.x % NSEG_;
  const size_t rowbase = (size_t)bh * S_ + (size_t)seg * SEGROWS;

  for (int i = tid; i < 128 * 16; i += 256) {
    const int d = i >> 4, n = i & 15;
    Wl[i] = (n < NF_) ? Wk[n * 128 + d] : 0.f;
  }

  const int ti0 = (w >> 2) * 2;  // d-tiles {ti0, ti0+1}
  const int tj  = w & 3;         // e-tile
  v8f c0 = {}; v8f c1 = {};

  for (int chunk = 0; chunk < SEGROWS / CH1_; ++chunk) {
    const size_t s0 = rowbase + (size_t)chunk * CH1_;

    // -------- cooperative global -> LDS (coalesced b128), V converted+transposed --------
    for (int i = tid; i < CH1_ * 16; i += 256) {
      const int r = i >> 4, c = (i & 15) * 4;
      const size_t g = (s0 + r) * HD_ + c;
      *(float4*)(Ktf + r * KSTRIDE + c) = *(const float4*)(Kt + g);
      *(float4*)(Kpf + r * KSTRIDE + c) = *(const float4*)(Kp + g);
      const float4 v = *(const float4*)(V + g);
      Bt[(c + 0) * TSTR1 + r] = (_Float16)v.x;
      Bt[(c + 1) * TSTR1 + r] = (_Float16)v.y;
      Bt[(c + 2) * TSTR1 + r] = (_Float16)v.z;
      Bt[(c + 3) * TSTR1 + r] = (_Float16)v.w;
    }
    if (chunk + 1 < SEGROWS / CH1_) {           // CDNA5 prefetch of next chunk
      const size_t ng = (s0 + CH1_) * HD_;
      __builtin_prefetch(Kt + ng + tid * 64, 0, 0);
      __builtin_prefetch(Kp + ng + tid * 64, 0, 0);
      __builtin_prefetch(V  + ng + tid * 64, 0, 0);
    }
    __syncthreads();

    // -------- per-row lambda_k + K_diff -> f16 transposed --------
    {
      const float* krow = Ktf + tid * KSTRIDE;
      const float* prow = Kpf + tid * KSTRIDE;
      const float lam = row_lambda(krow, prow, Wl, lamg);
      for (int d4 = 0; d4 < 16; ++d4) {
        const float4 a = *(const float4*)(krow + d4 * 4);
        const float4 b = *(const float4*)(prow + d4 * 4);
        At[(d4 * 4 + 0) * TSTR1 + tid] = (_Float16)(a.x - lam * b.x);
        At[(d4 * 4 + 1) * TSTR1 + tid] = (_Float16)(a.y - lam * b.y);
        At[(d4 * 4 + 2) * TSTR1 + tid] = (_Float16)(a.z - lam * b.z);
        At[(d4 * 4 + 3) * TSTR1 + tid] = (_Float16)(a.w - lam * b.w);
      }
    }
    __syncthreads();

    // -------- WMMA accumulate over this chunk's K=256 --------
#pragma unroll
    for (int ks = 0; ks < CH1_ / 32; ++ks) {
      v16h a0 = load_frag(At, TSTR1, ti0 * 16,        ks * 32, lane);
      v16h a1 = load_frag(At, TSTR1, (ti0 + 1) * 16,  ks * 32, lane);
      v16h b  = load_frag(Bt, TSTR1, tj * 16,         ks * 32, lane);
      c0 = wmma_f16(a0, b, c0);
      c1 = wmma_f16(a1, b, c1);
    }
    __syncthreads();
  }

  // -------- reduce partial 16x16 tiles into global KV --------
  float* kvb = KV + (size_t)bh * HD_ * HD_;
  const int n  = tj * 16 + (lane & 15);
  const int mh = (lane >> 4) * 8;     // C layout: VGPR g -> M=g (lanes<16) / g+8
#pragma unroll
  for (int g = 0; g < 8; ++g) {
    atomicAdd(kvb + (ti0 * 16 + mh + g) * HD_ + n,       c0[g]);
    atomicAdd(kvb + ((ti0 + 1) * 16 + mh + g) * HD_ + n, c1[g]);
  }
}

// ================= Kernel 2: O = (Q_t - lam_q Q'_t) * KV =================
extern "C" __global__ __launch_bounds__(256)
void td_o_kernel(const float* __restrict__ Qt, const float* __restrict__ Qp,
                 const float* __restrict__ lamg, const float* __restrict__ Wq,
                 const float* __restrict__ KV, float* __restrict__ O) {
  extern __shared__ char smem[];
  float*    Wl  = (float*)smem;
  float*    Qtf = (float*)(smem + LDS_W);
  float*    Qpf = (float*)(smem + LDS_W + LDS_KF);
  _Float16* Ah  = (_Float16*)(smem + LDS_W + 2 * LDS_KF);            // Q_diff [s][d]
  _Float16* Bt  = (_Float16*)(smem + LDS_W + 2 * LDS_KF + LDS_A2);   // KV^T   [e][d]

  const int tid  = threadIdx.x;
  const int lane = tid & 31;
  const int w    = tid >> 5;
  const int bh   = blockIdx.x >> 4;   // / (S_/ROWS2_)
  const int blk  = blockIdx.x & 15;
  const size_t rowbase = (size_t)bh * S_ + (size_t)blk * ROWS2_;

  for (int i = tid; i < 128 * 16; i += 256) {
    const int d = i >> 4, n = i & 15;
    Wl[i] = (n < NF_) ? Wq[n * 128 + d] : 0.f;
  }
  {  // KV -> f16 transposed [e][d]
    const float* kvb = KV + (size_t)bh * HD_ * HD_;
    for (int i = tid; i < HD_ * HD_; i += 256) {
      const int d = i >> 6, e = i & 63;
      Bt[e * BSTR2 + d] = (_Float16)kvb[i];
    }
  }
  for (int i = tid; i < ROWS2_ * 16; i += 256) {
    const int r = i >> 4, c = (i & 15) * 4;
    const size_t g = (rowbase + r) * HD_ + c;
    *(float4*)(Qtf + r * KSTRIDE + c) = *(const float4*)(Qt + g);
    *(float4*)(Qpf + r * KSTRIDE + c) = *(const float4*)(Qp + g);
  }
  __syncthreads();

  {  // per-row lambda_q + Q_diff -> f16 row-major (A is naturally [s][d])
    const float* qrow = Qtf + tid * KSTRIDE;
    const float* prow = Qpf + tid * KSTRIDE;
    const float lam = row_lambda(qrow, prow, Wl, lamg);
    _Float16* arow = Ah + tid * ASTR2;
    for (int d4 = 0; d4 < 16; ++d4) {
      const float4 a = *(const float4*)(qrow + d4 * 4);
      const float4 b = *(const float4*)(prow + d4 * 4);
      v4h h = { (_Float16)(a.x - lam * b.x), (_Float16)(a.y - lam * b.y),
                (_Float16)(a.z - lam * b.z), (_Float16)(a.w - lam * b.w) };
      *(v4h*)(arow + d4 * 4) = h;
    }
  }
  __syncthreads();

  // each wave: row-tiles {2w, 2w+1} x all 4 col-tiles; K=64 -> 2 k-steps
  v8f acc[2][4] = {};
#pragma unroll
  for (int ks = 0; ks < 2; ++ks) {
    v16h a0 = load_frag(Ah, ASTR2, (w * 2) * 16,     ks * 32, lane);
    v16h a1 = load_frag(Ah, ASTR2, (w * 2 + 1) * 16, ks * 32, lane);
#pragma unroll
    for (int t = 0; t < 4; ++t) {
      v16h b = load_frag(Bt, BSTR2, t * 16, ks * 32, lane);
      acc[0][t] = wmma_f16(a0, b, acc[0][t]);
      acc[1][t] = wmma_f16(a1, b, acc[1][t]);
    }
  }

  float* ob = O + rowbase * HD_;
  const int e0 = lane & 15;
  const int mh = (lane >> 4) * 8;
#pragma unroll
  for (int r = 0; r < 2; ++r)
#pragma unroll
    for (int t = 0; t < 4; ++t)
#pragma unroll
      for (int g = 0; g < 8; ++g) {
        const int srow = (w * 2 + r) * 16 + mh + g;
        ob[(size_t)srow * HD_ + t * 16 + e0] = acc[r][t][g];
      }
}

// ================= host =================
extern "C" void kernel_launch(void* const* d_in, const int* in_sizes, int n_in,
                              void* d_out, int out_size, void* d_ws, size_t ws_size,
                              hipStream_t stream) {
  (void)in_sizes; (void)n_in; (void)out_size; (void)ws_size;
  const float* Qt   = (const float*)d_in[0];
  const float* Qp   = (const float*)d_in[1];
  const float* Kt   = (const float*)d_in[2];
  const float* Kp   = (const float*)d_in[3];
  const float* V    = (const float*)d_in[4];
  const float* lamg = (const float*)d_in[5];
  const float* Wq   = (const float*)d_in[6];
  const float* Wk   = (const float*)d_in[7];
  float* O  = (float*)d_out;
  float* KV = (float*)d_ws;                        // BH*64*64 f32 = 1 MB scratch

  const int kvN = BH_ * HD_ * HD_;
  td_zero_kernel<<<(kvN + 255) / 256, 256, 0, stream>>>(KV, kvN);
  td_kv_kernel<<<BH_ * NSEG_, 256, LDS1_SZ, stream>>>(Kt, Kp, V, lamg, Wk, KV);
  td_o_kernel<<<BH_ * (S_ / ROWS2_), 256, LDS2_SZ, stream>>>(Qt, Qp, lamg, Wq, KV, O);
}
